// AttentionBlock_44143673868950
// MI455X (gfx1250) — matrix operand order repaired and checked by reading, not compiled, not run on hardware
//
#include <hip/hip_runtime.h>
#include <hip/hip_bf16.h>

typedef __attribute__((ext_vector_type(16))) _Float16 v16h;
typedef __attribute__((ext_vector_type(8)))  _Float16 v8h;
typedef __attribute__((ext_vector_type(8)))  float    v8f;

#define B_   32
#define C_   256
#define HW_  1024
#define NTOK (B_ * HW_)   // 32768
#define G_   8
#define CPG  32

#if defined(__has_builtin)
#if __has_builtin(__builtin_amdgcn_tensor_load_to_lds) && __has_builtin(__builtin_amdgcn_s_wait_tensorcnt)
#define HAVE_TDM 1
#endif
#endif

// ---------------- WMMA helpers (CDNA5 16x16x32 f16 -> f32) ----------------

__device__ __forceinline__ v8f wmma16(v16h a, v16h b, v8f c) {
    return __builtin_amdgcn_wmma_f32_16x16x32_f16(false, a, false, b, (short)0, c,
                                                  false, false);
}

// A-operand (16x32, M x K): lane%16 = M; per-lane K = e + (hi-lane ? 16 : 0)
__device__ __forceinline__ v16h frag_a(const _Float16* rowk, int lane) {
    const int khi = (lane & 16) ? 16 : 0;
    v8h lo = *(const v8h*)(rowk + khi);
    v8h hi = *(const v8h*)(rowk + khi + 8);
    v16h r;
#pragma unroll
    for (int i = 0; i < 8; ++i) { r[i] = lo[i]; r[i + 8] = hi[i]; }
    return r;
}

// B-operand (32x16, K x N) fed as B^T rows (N-major, K-contiguous):
// lane%16 = N; lane-group selects contiguous K block of 16.
__device__ __forceinline__ v16h frag_b(const _Float16* rowk, int lane) {
    const int kb = (lane & 16) ? 16 : 0;
    v8h lo = *(const v8h*)(rowk + kb);
    v8h hi = *(const v8h*)(rowk + kb + 8);
    v16h r;
#pragma unroll
    for (int i = 0; i < 8; ++i) { r[i] = lo[i]; r[i + 8] = hi[i]; }
    return r;
}

// ---------------- Tensor Data Mover: 2-D f16 tile, global -> LDS ----------------

#ifdef HAVE_TDM
typedef __attribute__((ext_vector_type(4))) unsigned int v4u;
typedef __attribute__((ext_vector_type(8))) int          v8i;
typedef __attribute__((ext_vector_type(4))) int          v4i;

__device__ __forceinline__ unsigned lds_off(const void* p) {
    return (unsigned)(uintptr_t)p;   // low 32 bits of flat addr = LDS offset
}

// D# per CDNA5 ISA: tile_h rows of tile_w f16 elems,
// source row stride row_stride (elems). LDS receives the tile densely packed.
__device__ __forceinline__ void tdm_load_2d_f16(unsigned lds_addr, const _Float16* gsrc,
                                                unsigned tile_w, unsigned tile_h,
                                                unsigned long long row_stride) {
    unsigned long long ga = (unsigned long long)(uintptr_t)gsrc;
    // group0: count=1 | lds_addr[63:32] | global_addr[120:64] | type=2 [127:126]
    unsigned long long q0 = 1ull | ((unsigned long long)lds_addr << 32);
    unsigned long long q1 = ga | (2ull << 62);
    v4u g0 = { (unsigned)q0, (unsigned)(q0 >> 32), (unsigned)q1, (unsigned)(q1 >> 32) };
    // group1: data_size=1 (2B) [17:16]; tensor_dim0 [79:48]; tensor_dim1 [111:80];
    //         tile_dim0 [127:112]; tile_dim1 [143:128]; tensor_dim0_stride [207:160]
    unsigned long long td0 = tile_w, td1 = tile_h;
    unsigned long long p0 = (1ull << 16) | (td0 << 48);
    unsigned long long p1 = (td0 >> 16) | (td1 << 16) | ((unsigned long long)tile_w << 48);
    unsigned long long p2 = (unsigned long long)tile_h | ((row_stride & 0xffffffffull) << 32);
    unsigned long long p3 = (row_stride >> 32);
    v8i g1 = { (int)(unsigned)p0, (int)(unsigned)(p0 >> 32),
               (int)(unsigned)p1, (int)(unsigned)(p1 >> 32),
               (int)(unsigned)p2, (int)(unsigned)(p2 >> 32),
               (int)(unsigned)p3, (int)(unsigned)(p3 >> 32) };
    v4i z4 = { 0, 0, 0, 0 };
    v8i z8 = { 0, 0, 0, 0, 0, 0, 0, 0 };
    // 6-arg form (clang-23 / therock-10.0 lane)
    __builtin_amdgcn_tensor_load_to_lds(g0, g1, z4, z4, z8, 0);
}
#define TDM_WAIT() __builtin_amdgcn_s_wait_tensorcnt(0)
#endif

// ---------------- 0) weight f32 -> f16 convert ----------------

__global__ void cvt_kernel(const float* __restrict__ qw, const float* __restrict__ ow,
                           _Float16* __restrict__ qwh, _Float16* __restrict__ owh) {
    int i = blockIdx.x * 256 + threadIdx.x;
    if (i < 768 * 256) qwh[i] = (_Float16)qw[i];
    if (i < 256 * 256) owh[i] = (_Float16)ow[i];
}

// ---------------- 1) GroupNorm -> token-major f16 ----------------

__global__ void gn_kernel(const float* __restrict__ x, const float* __restrict__ gamma,
                          const float* __restrict__ beta, _Float16* __restrict__ xn) {
    const int bg = blockIdx.x;
    const int b = bg >> 3, g = bg & 7;
    const int t = threadIdx.x;
    __shared__ float s_sum[256], s_sq[256];
    const int base = (b * C_ + g * CPG) * HW_;
    float sum = 0.f, sq = 0.f;
#pragma unroll 4
    for (int i = 0; i < 128; ++i) {
        float v = x[base + t + i * 256];
        sum += v; sq += v * v;
    }
    s_sum[t] = sum; s_sq[t] = sq;
    __syncthreads();
    for (int off = 128; off > 0; off >>= 1) {
        if (t < off) { s_sum[t] += s_sum[t + off]; s_sq[t] += s_sq[t + off]; }
        __syncthreads();
    }
    const float mean = s_sum[0] * (1.f / 32768.f);
    const float var  = s_sq[0] * (1.f / 32768.f) - mean * mean;
    const float rstd = rsqrtf(var + 1e-5f);
    for (int i = 0; i < 128; ++i) {
        int e = t + i * 256;
        int ch = e >> 10, s = e & 1023;
        int cg = g * CPG + ch;
        float v = (x[base + e] - mean) * rstd * gamma[cg] + beta[cg];
        xn[(b * HW_ + s) * C_ + cg] = (_Float16)v;
    }
}

// ---------------- GEMM core: 64x64 tile, TDM double-buffered K loop ----------------
// A: [M][256] f16, Bw: [N][256] f16 (B^T rows). acc: 4 x (16x16) per wave.

__device__ __forceinline__ void gemm_tile(const _Float16* __restrict__ A,
                                          const _Float16* __restrict__ Bw,
                                          _Float16 (*As)[64 * 32], _Float16 (*Bs)[64 * 32],
                                          int mb, int nb, int t, int w, int lane, int ln,
                                          v8f acc[4]) {
#ifdef HAVE_TDM
    if (w == 0) {
        tdm_load_2d_f16(lds_off(As[0]), A  + (size_t)mb * 256, 32, 64, 256);
        tdm_load_2d_f16(lds_off(Bs[0]), Bw + (size_t)nb * 256, 32, 64, 256);
    }
#pragma unroll
    for (int kk = 0; kk < 8; ++kk) {
        if (w == 0) TDM_WAIT();          // tile kk landed in LDS
        __syncthreads();                 // publish; also: readers of buf kk+1 done (iter kk-1)
        if (w == 0 && kk < 7) {          // prefetch tile kk+1 into the other buffer
            int k0n = (kk + 1) * 32;
            tdm_load_2d_f16(lds_off(As[(kk + 1) & 1]), A  + (size_t)mb * 256 + k0n, 32, 64, 256);
            tdm_load_2d_f16(lds_off(Bs[(kk + 1) & 1]), Bw + (size_t)nb * 256 + k0n, 32, 64, 256);
        }
        const _Float16* Ab = As[kk & 1];
        const _Float16* Bb = Bs[kk & 1];
        v16h a = frag_a(Ab + (w * 16 + ln) * 32, lane);
#pragma unroll
        for (int sub = 0; sub < 4; ++sub) {
            v16h bf = frag_b(Bb + (sub * 16 + ln) * 32, lane);
            acc[sub] = wmma16(a, bf, acc[sub]);
        }
        __syncthreads();                 // all reads of tile kk done
    }
#else
    for (int k0 = 0; k0 < 256; k0 += 32) {
#pragma unroll
        for (int i = 0; i < 2; ++i) {
            int c = t + i * 128;
            *(v8h*)(As[0] + c * 8) = *(const v8h*)(A  + (mb + (c >> 2)) * 256 + k0 + (c & 3) * 8);
            *(v8h*)(Bs[0] + c * 8) = *(const v8h*)(Bw + (nb + (c >> 2)) * 256 + k0 + (c & 3) * 8);
        }
        __syncthreads();
        v16h a = frag_a(As[0] + (w * 16 + ln) * 32, lane);
#pragma unroll
        for (int sub = 0; sub < 4; ++sub) {
            v16h bf = frag_b(Bs[0] + (sub * 16 + ln) * 32, lane);
            acc[sub] = wmma16(a, bf, acc[sub]);
        }
        __syncthreads();
    }
#endif
}

// ---------------- 2) QKV GEMM: [32768,256] x [768,256]^T -> f16 [32768,768] ----------------

__global__ void qkv_gemm(const _Float16* __restrict__ A, const _Float16* __restrict__ Bw,
                         const float* __restrict__ bias, _Float16* __restrict__ out) {
    __shared__ __align__(16) _Float16 As[2][64 * 32];
    __shared__ __align__(16) _Float16 Bs[2][64 * 32];
    const int t = threadIdx.x, w = t >> 5, lane = t & 31, ln = lane & 15;
    const int mb = blockIdx.x * 64, nb = blockIdx.y * 64;
    v8f acc[4] = {};
    gemm_tile(A, Bw, As, Bs, mb, nb, t, w, lane, ln, acc);
    const int rhi = (lane & 16) ? 8 : 0;
#pragma unroll
    for (int sub = 0; sub < 4; ++sub) {
        int d = nb + sub * 16 + ln;
        float bv = bias[d];
#pragma unroll
        for (int r = 0; r < 8; ++r) {
            int row = mb + w * 16 + r + rhi;
            out[row * 768 + d] = (_Float16)(acc[sub][r] + bv);
        }
    }
}

// ---------------- 3) flash attention per (batch, 64-query tile) ----------------

__global__ void attn_kernel(const _Float16* __restrict__ qkv, _Float16* __restrict__ ao) {
    extern __shared__ _Float16 smem[];
    _Float16* Qs  = smem;               // 64 x 256
    _Float16* Ks  = Qs + 64 * 256;      // 64 x 256
    _Float16* VTs = Ks + 64 * 256;      // 256 x 64 (transposed V)
    _Float16* Ps  = VTs + 256 * 64;     // 4 waves x 16 x 64

    const int t = threadIdx.x, w = t >> 5, lane = t & 31, ln = lane & 15;
    const int rhi = (lane & 16) ? 8 : 0;
    const int b = blockIdx.y, qt = blockIdx.x;
    const int qr0 = b * HW_ + qt * 64;

    // ---- load Q tile (cols 0..255) ----
#ifdef HAVE_TDM
    if (w == 0) {
        tdm_load_2d_f16(lds_off(Qs), qkv + (size_t)qr0 * 768, 256, 64, 768);
        TDM_WAIT();
    }
#else
#pragma unroll
    for (int i = 0; i < 16; ++i) {
        int c = t + i * 128;
        *(v8h*)(Qs + c * 8) = *(const v8h*)(qkv + (qr0 + (c >> 5)) * 768 + (c & 31) * 8);
    }
#endif
    __syncthreads();
    v16h qa[8];
#pragma unroll
    for (int kk = 0; kk < 8; ++kk)
        qa[kk] = frag_a(Qs + (w * 16 + ln) * 256 + kk * 32, lane);

    v8f o[16] = {};
    float mrow[8], lrow[8];
#pragma unroll
    for (int r = 0; r < 8; ++r) { mrow[r] = -1e30f; lrow[r] = 0.f; }

    for (int j = 0; j < 16; ++j) {
        __syncthreads();                 // prior tile reads done
        const int kr0 = b * HW_ + j * 64;
#ifdef HAVE_TDM
        if (w == 0)                      // K tile via TDM (overlaps V transpose below)
            tdm_load_2d_f16(lds_off(Ks), qkv + (size_t)kr0 * 768 + 256, 256, 64, 768);
#pragma unroll
        for (int i = 0; i < 16; ++i) {   // V tile: manual transposed copy
            int c = t + i * 128;
            int nloc = c >> 5, cb = (c & 31) * 8;
            v8h vv = *(const v8h*)(qkv + (kr0 + nloc) * 768 + 512 + cb);
#pragma unroll
            for (int q = 0; q < 8; ++q) VTs[(cb + q) * 64 + nloc] = vv[q];
        }
        if (w == 0) TDM_WAIT();
#else
#pragma unroll
        for (int i = 0; i < 16; ++i) {
            int c = t + i * 128;
            int nloc = c >> 5, cb = (c & 31) * 8;
            *(v8h*)(Ks + c * 8) = *(const v8h*)(qkv + (kr0 + nloc) * 768 + 256 + cb);
            v8h vv = *(const v8h*)(qkv + (kr0 + nloc) * 768 + 512 + cb);
#pragma unroll
            for (int q = 0; q < 8; ++q) VTs[(cb + q) * 64 + nloc] = vv[q];
        }
#endif
        __syncthreads();

        v8f s[4] = {};                   // S = Q * K^T  (16 x 64 per wave)
#pragma unroll
        for (int kk = 0; kk < 8; ++kk)
#pragma unroll
            for (int sub = 0; sub < 4; ++sub) {
                v16h kb = frag_b(Ks + (sub * 16 + ln) * 256 + kk * 32, lane);
                s[sub] = wmma16(qa[kk], kb, s[sub]);
            }
#pragma unroll
        for (int sub = 0; sub < 4; ++sub)
#pragma unroll
            for (int r = 0; r < 8; ++r) s[sub][r] *= 0.0625f;   // 1/sqrt(256)

        float mnew[8], alpha[8];
#pragma unroll
        for (int r = 0; r < 8; ++r) {    // row reduction within each 16-lane group
            float v = fmaxf(fmaxf(s[0][r], s[1][r]), fmaxf(s[2][r], s[3][r]));
            v = fmaxf(v, __shfl_xor(v, 8, 32));
            v = fmaxf(v, __shfl_xor(v, 4, 32));
            v = fmaxf(v, __shfl_xor(v, 2, 32));
            v = fmaxf(v, __shfl_xor(v, 1, 32));
            mnew[r] = fmaxf(mrow[r], v);
            alpha[r] = __expf(mrow[r] - mnew[r]);
            mrow[r] = mnew[r];
        }
#pragma unroll
        for (int sub = 0; sub < 4; ++sub)
#pragma unroll
            for (int r = 0; r < 8; ++r) s[sub][r] = __expf(s[sub][r] - mnew[r]);
#pragma unroll
        for (int r = 0; r < 8; ++r) {
            float rs = s[0][r] + s[1][r] + s[2][r] + s[3][r];
            rs += __shfl_xor(rs, 8, 32);
            rs += __shfl_xor(rs, 4, 32);
            rs += __shfl_xor(rs, 2, 32);
            rs += __shfl_xor(rs, 1, 32);
            lrow[r] = lrow[r] * alpha[r] + rs;
        }
#pragma unroll
        for (int ct = 0; ct < 16; ++ct)
#pragma unroll
            for (int r = 0; r < 8; ++r) o[ct][r] *= alpha[r];

        // C-layout P -> LDS -> A-layout fragments (per-wave region)
#pragma unroll
        for (int sub = 0; sub < 4; ++sub)
#pragma unroll
            for (int r = 0; r < 8; ++r)
                Ps[w * 1024 + (r + rhi) * 64 + sub * 16 + ln] = (_Float16)s[sub][r];
        __syncthreads();

#pragma unroll
        for (int kk2 = 0; kk2 < 2; ++kk2) {   // O += P * V
            v16h pa = frag_a(Ps + w * 1024 + ln * 64 + kk2 * 32, lane);
#pragma unroll
            for (int ct = 0; ct < 16; ++ct) {
                v16h vb = frag_b(VTs + (ct * 16 + ln) * 64 + kk2 * 32, lane);
                o[ct] = wmma16(pa, vb, o[ct]);
            }
        }
    }

    float linv[8];
#pragma unroll
    for (int r = 0; r < 8; ++r) linv[r] = 1.f / lrow[r];
#pragma unroll
    for (int ct = 0; ct < 16; ++ct)
#pragma unroll
        for (int r = 0; r < 8; ++r) {
            int row = qr0 + w * 16 + r + rhi;
            ao[row * C_ + ct * 16 + ln] = (_Float16)(o[ct][r] * linv[r]);
        }
}

// ---------------- 4) out-proj GEMM + bias + residual, scatter to (b,c,h,w) ----------------

__global__ void outproj_gemm(const _Float16* __restrict__ A, const _Float16* __restrict__ Bw,
                             const float* __restrict__ bias, const float* __restrict__ xres,
                             float* __restrict__ out) {
    __shared__ __align__(16) _Float16 As[2][64 * 32];
    __shared__ __align__(16) _Float16 Bs[2][64 * 32];
    const int t = threadIdx.x, w = t >> 5, lane = t & 31, ln = lane & 15;
    const int mb = blockIdx.x * 64, nb = blockIdx.y * 64;
    v8f acc[4] = {};
    gemm_tile(A, Bw, As, Bs, mb, nb, t, w, lane, ln, acc);
    const int rhi = (lane & 16) ? 8 : 0;
#pragma unroll
    for (int sub = 0; sub < 4; ++sub) {
        int d = nb + sub * 16 + ln;
        float bv = bias[d];
#pragma unroll
        for (int r = 0; r < 8; ++r) {
            int row = mb + w * 16 + r + rhi;       // global token row
            int bb = row >> 10, sidx = row & 1023;
            int oi = (bb * C_ + d) * HW_ + sidx;
            out[oi] = acc[sub][r] + bv + xres[oi];
        }
    }
}

// ---------------- host launch ----------------

extern "C" void kernel_launch(void* const* d_in, const int* in_sizes, int n_in,
                              void* d_out, int out_size, void* d_ws, size_t ws_size,
                              hipStream_t stream) {
    const float* x        = (const float*)d_in[0];
    const float* gn_gamma = (const float*)d_in[1];
    const float* gn_beta  = (const float*)d_in[2];
    const float* qkv_w    = (const float*)d_in[3];
    const float* qkv_b    = (const float*)d_in[4];
    const float* out_w    = (const float*)d_in[5];
    const float* out_b    = (const float*)d_in[6];
    float* out = (float*)d_out;

    // workspace layout (f16): xn | qkv | attn_out | qkv_w_h | out_w_h  (~84.4 MB)
    _Float16* xn   = (_Float16*)d_ws;
    _Float16* qkvh = xn   + (size_t)NTOK * C_;
    _Float16* aoh  = qkvh + (size_t)NTOK * 768;
    _Float16* qwh  = aoh  + (size_t)NTOK * C_;
    _Float16* owh  = qwh  + (size_t)768 * 256;

    const size_t attn_lds = (size_t)(64 * 256 * 3 + 4 * 16 * 64) * sizeof(_Float16); // 106496 B
    (void)hipFuncSetAttribute((const void*)attn_kernel,
                              hipFuncAttributeMaxDynamicSharedMemorySize, (int)attn_lds);

    cvt_kernel<<<768, 256, 0, stream>>>(qkv_w, out_w, qwh, owh);
    gn_kernel<<<B_ * G_, 256, 0, stream>>>(x, gn_gamma, gn_beta, xn);
    qkv_gemm<<<dim3(NTOK / 64, 12), 128, 0, stream>>>(xn, qwh, qkv_b, qkvh);
    attn_kernel<<<dim3(HW_ / 64, B_), 128, attn_lds, stream>>>(qkvh, aoh);
    outproj_gemm<<<dim3(NTOK / 64, 4), 128, 0, stream>>>(aoh, owh, out_b, x, out);
}
